// Decoder_64132451663924
// MI455X (gfx1250) — matrix operand order repaired and checked by reading, not compiled, not run on hardware
//
#include <hip/hip_runtime.h>
#include <hip/hip_bf16.h>
#include <stdint.h>

typedef __attribute__((ext_vector_type(16))) __bf16 v16bf;
typedef __attribute__((ext_vector_type(8)))  float  v8f;

// ---------------- problem constants ----------------
constexpr int BATCH = 32;
constexpr int TENC  = 400;
constexpr int TMEL  = 400;
constexpr int NMELS = 80;
constexpr int PREN  = 256;
constexpr int HID   = 1024;
constexpr int ENC   = 512;
constexpr int ATTD  = 128;
constexpr int NF    = 32;
constexpr int KSZ   = 31;
constexpr int GATES = 4096;
// fused K dims: x_att = [p(256) | ctx(512) | ah(1024)] ; x_dec = [ah(1024) | ctx(512) | dh(1024)]
constexpr int KATT  = 1792;
constexpr int KDEC  = 2560;
constexpr int KT_ATT = KATT / 32;   // 56
constexpr int KT_DEC = KDEC / 32;   // 80
constexpr int NT     = GATES / 16;  // 256
constexpr int MTT    = BATCH / 16;  // 2
constexpr int NWG    = 64;
constexpr int BLK    = 256;

// ---------------- workspace layout (bytes) ----------------
constexpr size_t SZ_WATT = (size_t)KT_ATT * NT * 32 * 16 * 2;
constexpr size_t SZ_WDEC = (size_t)KT_DEC * NT * 32 * 16 * 2;
constexpr size_t SZ_XATT = (size_t)KT_ATT * MTT * 32 * 16 * 2;
constexpr size_t SZ_XDEC = (size_t)KT_DEC * MTT * 32 * 16 * 2;
constexpr size_t SZ_PRE  = (size_t)TMEL * BATCH * PREN * 4;
constexpr size_t SZ_PM   = (size_t)BATCH * TENC * ATTD * 4;
constexpr size_t SZ_G    = (size_t)BATCH * GATES * 4;

constexpr size_t OFF_WATT = 0;
constexpr size_t OFF_WDEC = OFF_WATT + SZ_WATT;
constexpr size_t OFF_XATT = OFF_WDEC + SZ_WDEC;
constexpr size_t OFF_XDEC = OFF_XATT + SZ_XATT;
constexpr size_t OFF_PRE  = OFF_XDEC + SZ_XDEC;
constexpr size_t OFF_PM   = OFF_PRE + SZ_PRE;
constexpr size_t OFF_GA   = OFF_PM + SZ_PM;
constexpr size_t OFF_GD   = OFF_GA + SZ_G;
constexpr size_t OFF_BAR  = OFF_GD + SZ_G;

__device__ __forceinline__ float sigm(float x) { return 1.0f / (1.0f + __expf(-x)); }

// Pack one scalar of an activation matrix [32 x K] into the CDNA5 A-fragment layout
// (16-bit A 16x32: lanes 0-15 -> K {0..7,16..23}, lanes 16-31 -> K {8..15,24..31}).
__device__ __forceinline__ void packA(__hip_bfloat16* frag, int b, int K, float v) {
  int kt = K >> 5, kk = K & 31;
  int mT = b >> 4, mr = b & 15;
  int lane, e;
  if (kk < 8)       { lane = mr;      e = kk;      }
  else if (kk < 16) { lane = mr + 16; e = kk - 8;  }
  else if (kk < 24) { lane = mr;      e = kk - 8;  }
  else              { lane = mr + 16; e = kk - 16; }
  frag[(((size_t)kt * MTT + mT) * 32 + lane) * 16 + e] = __float2bfloat16(v);
}

// ---------------- precompute kernels ----------------

__global__ void k_prenet(const float* __restrict__ mel, const float* __restrict__ W1,
                         const float* __restrict__ W2, float* __restrict__ pre) {
  __shared__ float s_in[BATCH * NMELS];
  __shared__ float s_p1[BATCH * PREN];
  int t = blockIdx.x, tid = threadIdx.x;
  for (int i = tid; i < BATCH * NMELS; i += BLK) {
    int b = i / NMELS, m = i % NMELS;
    s_in[i] = (t == 0) ? 0.0f : mel[(size_t)b * NMELS * TMEL + (size_t)m * TMEL + (t - 1)];
  }
  __syncthreads();
  for (int i = tid; i < BATCH * PREN; i += BLK) {
    int b = i / PREN, j = i % PREN;
    float s = 0.0f;
    for (int m = 0; m < NMELS; ++m) s += s_in[b * NMELS + m] * W1[m * PREN + j];
    s_p1[i] = fmaxf(s, 0.0f);
  }
  __syncthreads();
  for (int i = tid; i < BATCH * PREN; i += BLK) {
    int b = i / PREN, j = i % PREN;
    float s = 0.0f;
    for (int k = 0; k < PREN; ++k) s += s_p1[b * PREN + k] * W2[k * PREN + j];
    pre[((size_t)t * BATCH + b) * PREN + j] = fmaxf(s, 0.0f);
  }
}

__global__ void k_pm(const float* __restrict__ mem, const float* __restrict__ memW,
                     float* __restrict__ pm) {
  size_t i = (size_t)blockIdx.x * BLK + threadIdx.x;
  if (i >= (size_t)BATCH * TENC * ATTD) return;
  int a = (int)(i % ATTD);
  size_t row = i / ATTD;
  const float* mr = mem + row * ENC;
  float s = 0.0f;
  for (int d = 0; d < ENC; ++d) s += mr[d] * memW[d * ATTD + a];
  pm[i] = s;
}

// bf16 B-fragments: lanes 0-15 -> K {0..7,16..23} ; lanes 16-31 -> K {8..15,24..31} (+kt*32) ; N = lane%16.
__global__ void k_packW(const float* __restrict__ aWih, const float* __restrict__ aWhh,
                        const float* __restrict__ dWih, const float* __restrict__ dWhh,
                        __hip_bfloat16* __restrict__ watt, __hip_bfloat16* __restrict__ wdec) {
  const size_t natt = (size_t)KT_ATT * NT * 512;
  const size_t ndec = (size_t)KT_DEC * NT * 512;
  for (size_t i = (size_t)blockIdx.x * BLK + threadIdx.x; i < natt + ndec;
       i += (size_t)gridDim.x * BLK) {
    bool isatt = i < natt;
    size_t idx = isatt ? i : i - natt;
    int KT = isatt ? KT_ATT : KT_DEC;
    int e = (int)(idx & 15);
    int lane = (int)((idx >> 4) & 31);
    int kt = (int)((idx >> 9) % KT);
    int nT = (int)(idx / (512ull * KT));
    int N = nT * 16 + (lane & 15);
    int K = kt * 32 + ((e & 8) << 1) + ((lane >> 4) << 3) + (e & 7);
    float v;
    if (isatt) v = (K < 768)  ? aWih[(size_t)K * GATES + N] : aWhh[(size_t)(K - 768) * GATES + N];
    else       v = (K < 1536) ? dWih[(size_t)K * GATES + N] : dWhh[(size_t)(K - 1536) * GATES + N];
    (isatt ? watt : wdec)[idx] = __float2bfloat16(v);
  }
}

__global__ void k_init(char* ws) {
  size_t gid = (size_t)blockIdx.x * BLK + threadIdx.x;
  size_t stride = (size_t)gridDim.x * BLK;
  uint32_t* xz = (uint32_t*)(ws + OFF_XATT);
  for (size_t i = gid; i < (SZ_XATT + SZ_XDEC) / 4; i += stride) xz[i] = 0u;
  if (gid == 0) { int* bar = (int*)(ws + OFF_BAR); bar[0] = 0; bar[1] = 0; }
}

__global__ void k_packP0(char* ws) {
  size_t gid = (size_t)blockIdx.x * BLK + threadIdx.x;
  if (gid >= (size_t)BATCH * PREN) return;
  int b = (int)(gid / PREN), j = (int)(gid % PREN);
  const float* pre = (const float*)(ws + OFF_PRE);
  packA((__hip_bfloat16*)(ws + OFF_XATT), b, j, pre[gid]);
}

// ---------------- persistent decoder ----------------

struct DecArgs {
  const float* memory;
  const int*   mlen;
  const float* wv;
  const float* convW;
  const float* denseW;
  const float* queryW;
  const float* att_b;
  const float* dec_b;
  const float* projW;
  const float* projb;
  const float* gateW;
  const float* gateb;
  float* out;
  char*  ws;
};

__device__ __forceinline__ void grid_sync(int* bar) {
  __threadfence();
  __syncthreads();
  if (threadIdx.x == 0) {
    int g = __hip_atomic_load(&bar[1], __ATOMIC_ACQUIRE, __HIP_MEMORY_SCOPE_AGENT);
    int a = __hip_atomic_fetch_add(&bar[0], 1, __ATOMIC_ACQ_REL, __HIP_MEMORY_SCOPE_AGENT);
    if (a == NWG - 1) {
      __hip_atomic_store(&bar[0], 0, __ATOMIC_RELAXED, __HIP_MEMORY_SCOPE_AGENT);
      __hip_atomic_fetch_add(&bar[1], 1, __ATOMIC_ACQ_REL, __HIP_MEMORY_SCOPE_AGENT);
    } else {
      while (__hip_atomic_load(&bar[1], __ATOMIC_ACQUIRE, __HIP_MEMORY_SCOPE_AGENT) == g)
        __builtin_amdgcn_s_sleep(1);
    }
  }
  __syncthreads();
  __threadfence();
}

__global__ void __launch_bounds__(BLK) k_decoder(DecArgs A) {
  // ---- per-WG resident state (one WG == one batch row for wg < 32) ----
  __shared__ float sred[BLK];
  __shared__ float s_pp[NMELS];
  __shared__ float s_aw[TENC];        // current attention weights
  __shared__ float s_awp[TENC];       // previous attention weights
  __shared__ float s_awc[TENC];       // cumulative attention weights
  __shared__ float s_e[TENC];         // energies
  __shared__ float s_q[ATTD];         // query
  __shared__ float s_ah[HID];         // attention-LSTM hidden
  __shared__ float s_dh[HID];         // decoder-LSTM hidden
  __shared__ float s_ctx[ENC];        // context
  __shared__ float s_dense[NF * ATTD];// location dense weights (staged once)
  __shared__ float s_conv[NF * 62];   // location conv weights (staged once)

  char* ws = A.ws;
  const int tid = threadIdx.x;
  const int wg = blockIdx.x;
  const int lane = tid & 31;
  int* bar = (int*)(ws + OFF_BAR);
  float* out_mel  = A.out;
  float* out_gate = A.out + (size_t)BATCH * NMELS * TMEL;
  float* out_al   = out_gate + (size_t)BATCH * TMEL;
  __hip_bfloat16* xatt = (__hip_bfloat16*)(ws + OFF_XATT);
  __hip_bfloat16* xdec = (__hip_bfloat16*)(ws + OFF_XDEC);
  const int b = wg;  // batch owned by this WG (valid when wg < 32)

  // cell states live in VGPRs: element j = ii*BLK + tid
  float acr[4] = {0.f, 0.f, 0.f, 0.f};
  float dcr[4] = {0.f, 0.f, 0.f, 0.f};
  int blen = 0;

  if (wg < BATCH) {
    for (int i = tid; i < NF * ATTD; i += BLK) s_dense[i] = A.denseW[i];
    for (int i = tid; i < NF * 62; i += BLK) s_conv[i] = A.convW[i];
    for (int i = tid; i < TENC; i += BLK) { s_awp[i] = 0.f; s_awc[i] = 0.f; }
    for (int i = tid; i < HID; i += BLK) s_dh[i] = 0.f;
    blen = A.mlen[b];
  }
  __syncthreads();

  for (int t = 0; t < TMEL; ++t) {
    // ================= Phase 1: att-GEMM (WGs 32..63) || dec-LSTM+proj of t-1 (WGs 0..31) ===
    if (wg >= BATCH) {
      // [32,1792] x [1792,4096]: 256 waves, each 1 mT x 2 nT with A-fragment reuse
      int lw = (wg - BATCH) * (BLK / 32) + (tid >> 5);  // 0..255
      int mT = lw & 1;
      int nT0 = (lw >> 1) * 2;
      const v16bf* xf = (const v16bf*)(ws + OFF_XATT);
      const v16bf* wf = (const v16bf*)(ws + OFF_WATT);
      v8f acc0 = {}, acc1 = {};
      for (int kt = 0; kt < KT_ATT; ++kt) {
        v16bf a  = xf[(kt * MTT + mT) * 32 + lane];
        v16bf b0 = wf[((size_t)nT0 * KT_ATT + kt) * 32 + lane];
        v16bf b1 = wf[((size_t)(nT0 + 1) * KT_ATT + kt) * 32 + lane];
        if (kt + 1 < KT_ATT)
          __builtin_prefetch((const void*)&wf[((size_t)nT0 * KT_ATT + kt + 1) * 32 + lane], 0, 3);
        acc0 = __builtin_amdgcn_wmma_f32_16x16x32_bf16(false, a, false, b0, (short)0, acc0, false, false);
        acc1 = __builtin_amdgcn_wmma_f32_16x16x32_bf16(false, a, false, b1, (short)0, acc1, false, false);
      }
      float* g = (float*)(ws + OFF_GA);
      int rb = (lane < 16) ? 0 : 8;
      int col0 = nT0 * 16 + (lane & 15);
#pragma unroll
      for (int r = 0; r < 8; ++r) {
        g[(size_t)(mT * 16 + rb + r) * GATES + col0] = acc0[r];
        g[(size_t)(mT * 16 + rb + r) * GATES + col0 + 16] = acc1[r];
      }
    } else if (t > 0) {
      // ---- dec-LSTM pointwise of step t-1 + projections ----
      const float* g = (const float*)(ws + OFF_GD) + (size_t)b * GATES;
#pragma unroll
      for (int ii = 0; ii < 4; ++ii) {
        int j = ii * BLK + tid;
        float ig = sigm(g[j] + A.dec_b[j]);
        float fg = sigm(g[HID + j] + A.dec_b[HID + j]);
        float gg = tanhf(g[2 * HID + j] + A.dec_b[2 * HID + j]);
        float og = sigm(g[3 * HID + j] + A.dec_b[3 * HID + j]);
        float c = fg * dcr[ii] + ig * gg;
        dcr[ii] = c;
        s_dh[j] = og * tanhf(c);
      }
      if (tid < NMELS) s_pp[tid] = 0.f;
      __syncthreads();
      // pack dh(t-1) into x_dec for step t
      for (int k = tid; k < HID; k += BLK) packA(xdec, b, 1536 + k, s_dh[k]);
      // mel projection: 3 partial threads per output channel
      if (tid < 240) {
        int m = tid / 3, part = tid % 3;
        float s = 0.f;
        for (int k = part * 512; k < part * 512 + 512; ++k) {
          float x = (k < HID) ? s_dh[k] : s_ctx[k - HID];
          s += x * A.projW[(size_t)k * NMELS + m];
        }
        atomicAdd(&s_pp[m], s);
      }
      // gate projection
      float p = 0.f;
      for (int k = tid; k < HID + ENC; k += BLK) {
        float x = (k < HID) ? s_dh[k] : s_ctx[k - HID];
        p += x * A.gateW[k];
      }
      sred[tid] = p; __syncthreads();
      for (int s = BLK / 2; s > 0; s >>= 1) {
        if (tid < s) sred[tid] += sred[tid + s];
        __syncthreads();
      }
      if (tid < NMELS)
        out_mel[(size_t)b * NMELS * TMEL + (size_t)tid * TMEL + (t - 1)] = s_pp[tid] + A.projb[tid];
      if (tid == 0) out_gate[(size_t)b * TMEL + (t - 1)] = sred[0] + A.gateb[0];
    }
    grid_sync(bar);

    // ================= Phase 2: full attention step for batch b (WGs 0..31) =================
    if (wg < BATCH) {
      // --- att-LSTM pointwise ---
      const float* g = (const float*)(ws + OFF_GA) + (size_t)b * GATES;
#pragma unroll
      for (int ii = 0; ii < 4; ++ii) {
        int j = ii * BLK + tid;
        float ig = sigm(g[j] + A.att_b[j]);
        float fg = sigm(g[HID + j] + A.att_b[HID + j]);
        float gg = tanhf(g[2 * HID + j] + A.att_b[2 * HID + j]);
        float og = sigm(g[3 * HID + j] + A.att_b[3 * HID + j]);
        float c = fg * acr[ii] + ig * gg;
        acr[ii] = c;
        s_ah[j] = og * tanhf(c);
      }
      __syncthreads();
      // --- query + ah packs ---
      if (tid < ATTD) {
        float s = 0.f;
        for (int k = 0; k < HID; ++k) s += s_ah[k] * A.queryW[k * ATTD + tid];
        s_q[tid] = s;
      }
      for (int k = tid; k < HID; k += BLK) {
        float h = s_ah[k];
        packA(xatt, b, 768 + k, h);  // x_att for step t+1
        packA(xdec, b, k, h);        // x_dec ah part for step t
      }
      __syncthreads();
      // --- location conv + energies ---
      for (int tp = tid; tp < TENC; tp += BLK) {
        float cf[NF];
#pragma unroll
        for (int f = 0; f < NF; ++f) cf[f] = 0.f;
        for (int k = 0; k < KSZ; ++k) {
          int tt = tp + k - 15;
          if (tt >= 0 && tt < TENC) {
            float a0 = s_awp[tt], a1 = s_awc[tt];
#pragma unroll
            for (int f = 0; f < NF; ++f)
              cf[f] += a0 * s_conv[f * 62 + k] + a1 * s_conv[f * 62 + 31 + k];
          }
        }
        const float* pmr = (const float*)(ws + OFF_PM) + ((size_t)b * TENC + tp) * ATTD;
        float e = 0.f;
        for (int a = 0; a < ATTD; ++a) {
          float loc = 0.f;
#pragma unroll
          for (int f = 0; f < NF; ++f) loc += cf[f] * s_dense[f * ATTD + a];
          e += tanhf(s_q[a] + pmr[a] + loc) * A.wv[a];
        }
        s_e[tp] = (tp >= blen) ? -3.0e38f : e;
      }
      __syncthreads();
      // --- softmax over T_enc ---
      float m = -3.4e38f;
      for (int t2 = tid; t2 < TENC; t2 += BLK) m = fmaxf(m, s_e[t2]);
      sred[tid] = m; __syncthreads();
      for (int s = BLK / 2; s > 0; s >>= 1) {
        if (tid < s) sred[tid] = fmaxf(sred[tid], sred[tid + s]);
        __syncthreads();
      }
      float smax = sred[0]; __syncthreads();
      float ssum = 0.f;
      for (int t2 = tid; t2 < TENC; t2 += BLK) {
        float x = __expf(s_e[t2] - smax);
        s_aw[t2] = x; ssum += x;
      }
      sred[tid] = ssum; __syncthreads();
      for (int s = BLK / 2; s > 0; s >>= 1) {
        if (tid < s) sred[tid] += sred[tid + s];
        __syncthreads();
      }
      float inv = 1.f / sred[0]; __syncthreads();
      for (int t2 = tid; t2 < TENC; t2 += BLK) {
        float w = s_aw[t2] * inv;
        s_aw[t2] = w; s_awp[t2] = w; s_awc[t2] += w;
        out_al[(size_t)b * TENC * TMEL + (size_t)t * TENC + t2] = w;
      }
      __syncthreads();
      // --- context = aw @ memory[b] ---
      for (int d = tid; d < ENC; d += BLK) {
        float c = 0.f;
        const float* mp = A.memory + (size_t)b * TENC * ENC + d;
        for (int t2 = 0; t2 < TENC; ++t2) c += s_aw[t2] * mp[(size_t)t2 * ENC];
        s_ctx[d] = c;
      }
      __syncthreads();
      // --- packs: ctx (step t dec GEMM + step t+1 att GEMM), prenet(t+1) ---
      for (int d = tid; d < ENC; d += BLK) {
        float c = s_ctx[d];
        packA(xdec, b, 1024 + d, c);
        packA(xatt, b, 256 + d, c);
      }
      if (t + 1 < TMEL) {
        const float* pr = (const float*)(ws + OFF_PRE) + ((size_t)(t + 1) * BATCH + b) * PREN;
        for (int j = tid; j < PREN; j += BLK) packA(xatt, b, j, pr[j]);
      }
    }
    grid_sync(bar);

    // ================= Phase 3: dec-GEMM [32,2560]x[2560,4096], all 512 waves ===============
    {
      int wid = wg * (BLK / 32) + (tid >> 5);
      int mT = wid & 1;
      int nT = wid >> 1;
      const v16bf* xf = (const v16bf*)(ws + OFF_XDEC);
      const v16bf* wf = (const v16bf*)(ws + OFF_WDEC);
      v8f acc = {};
      for (int kt = 0; kt < KT_DEC; ++kt) {
        v16bf a = xf[(kt * MTT + mT) * 32 + lane];
        v16bf bb = wf[((size_t)nT * KT_DEC + kt) * 32 + lane];
        if (kt + 1 < KT_DEC)
          __builtin_prefetch((const void*)&wf[((size_t)nT * KT_DEC + kt + 1) * 32 + lane], 0, 3);
        acc = __builtin_amdgcn_wmma_f32_16x16x32_bf16(false, a, false, bb, (short)0, acc, false, false);
      }
      float* g = (float*)(ws + OFF_GD);
      int col = nT * 16 + (lane & 15);
      int rb = (lane < 16) ? 0 : 8;
#pragma unroll
      for (int r = 0; r < 8; ++r)
        g[(size_t)(mT * 16 + rb + r) * GATES + col] = acc[r];
    }
    grid_sync(bar);
  }

  // ================= Epilogue: dec-LSTM + projections for final step =================
  if (wg < BATCH) {
    const float* g = (const float*)(ws + OFF_GD) + (size_t)b * GATES;
#pragma unroll
    for (int ii = 0; ii < 4; ++ii) {
      int j = ii * BLK + tid;
      float ig = sigm(g[j] + A.dec_b[j]);
      float fg = sigm(g[HID + j] + A.dec_b[HID + j]);
      float gg = tanhf(g[2 * HID + j] + A.dec_b[2 * HID + j]);
      float og = sigm(g[3 * HID + j] + A.dec_b[3 * HID + j]);
      float c = fg * dcr[ii] + ig * gg;
      s_dh[j] = og * tanhf(c);
    }
    if (tid < NMELS) s_pp[tid] = 0.f;
    __syncthreads();
    if (tid < 240) {
      int m = tid / 3, part = tid % 3;
      float s = 0.f;
      for (int k = part * 512; k < part * 512 + 512; ++k) {
        float x = (k < HID) ? s_dh[k] : s_ctx[k - HID];
        s += x * A.projW[(size_t)k * NMELS + m];
      }
      atomicAdd(&s_pp[m], s);
    }
    float p = 0.f;
    for (int k = tid; k < HID + ENC; k += BLK) {
      float x = (k < HID) ? s_dh[k] : s_ctx[k - HID];
      p += x * A.gateW[k];
    }
    sred[tid] = p; __syncthreads();
    for (int s = BLK / 2; s > 0; s >>= 1) {
      if (tid < s) sred[tid] += sred[tid + s];
      __syncthreads();
    }
    if (tid < NMELS)
      out_mel[(size_t)b * NMELS * TMEL + (size_t)tid * TMEL + (TMEL - 1)] = s_pp[tid] + A.projb[tid];
    if (tid == 0) out_gate[(size_t)b * TMEL + (TMEL - 1)] = sred[0] + A.gateb[0];
  }
}

// ---------------- host launcher ----------------
extern "C" void kernel_launch(void* const* d_in, const int* in_sizes, int n_in,
                              void* d_out, int out_size, void* d_ws, size_t ws_size,
                              hipStream_t stream) {
  const float* memory     = (const float*)d_in[0];
  const float* mel_target = (const float*)d_in[1];
  const int*   mlen       = (const int*)d_in[2];
  const float* prenet_W1  = (const float*)d_in[3];
  const float* prenet_W2  = (const float*)d_in[4];
  const float* query_W    = (const float*)d_in[5];
  const float* memory_W   = (const float*)d_in[6];
  const float* weight_vec = (const float*)d_in[7];
  const float* loc_conv_W = (const float*)d_in[8];
  const float* loc_dense_W= (const float*)d_in[9];
  const float* att_Wih    = (const float*)d_in[10];
  const float* att_Whh    = (const float*)d_in[11];
  const float* att_b      = (const float*)d_in[12];
  const float* dec_Wih    = (const float*)d_in[13];
  const float* dec_Whh    = (const float*)d_in[14];
  const float* dec_b      = (const float*)d_in[15];
  const float* proj_W     = (const float*)d_in[16];
  const float* proj_b     = (const float*)d_in[17];
  const float* gate_W     = (const float*)d_in[18];
  const float* gate_b     = (const float*)d_in[19];
  (void)in_sizes; (void)n_in; (void)out_size; (void)ws_size;

  char* ws = (char*)d_ws;

  k_prenet<<<TMEL, BLK, 0, stream>>>(mel_target, prenet_W1, prenet_W2,
                                     (float*)(ws + OFF_PRE));
  k_pm<<<(BATCH * TENC * ATTD + BLK - 1) / BLK, BLK, 0, stream>>>(
      memory, memory_W, (float*)(ws + OFF_PM));
  k_packW<<<4096, BLK, 0, stream>>>(att_Wih, att_Whh, dec_Wih, dec_Whh,
                                    (__hip_bfloat16*)(ws + OFF_WATT),
                                    (__hip_bfloat16*)(ws + OFF_WDEC));
  k_init<<<256, BLK, 0, stream>>>(ws);
  k_packP0<<<(BATCH * PREN + BLK - 1) / BLK, BLK, 0, stream>>>(ws);

  DecArgs A;
  A.memory = memory; A.mlen = mlen; A.wv = weight_vec;
  A.convW = loc_conv_W; A.denseW = loc_dense_W; A.queryW = query_W;
  A.att_b = att_b; A.dec_b = dec_b;
  A.projW = proj_W; A.projb = proj_b; A.gateW = gate_W; A.gateb = gate_b;
  A.out = (float*)d_out; A.ws = ws;
  k_decoder<<<NWG, BLK, 0, stream>>>(A);
}
